// TemporalEncoder_40604620816479
// MI455X (gfx1250) — matrix operand order repaired and checked by reading, not compiled, not run on hardware
//
#include <hip/hip_runtime.h>
#include <hip/hip_bf16.h>

#define DEVINL __device__ __forceinline__

typedef __attribute__((ext_vector_type(16))) __bf16 v16bf;
typedef __attribute__((ext_vector_type(8)))  __bf16 v8bf;
typedef __attribute__((ext_vector_type(8)))  float  v8f;

static constexpr int B_  = 16;
static constexpr int T_  = 2048;
static constexpr int C_  = 256;
static constexpr int IC_ = 128;

DEVINL v8f wmma_bf(v16bf a, v16bf b, v8f c) {
  // v_wmma_f32_16x16x32_bf16: D = A(16x32) * B(32x16) + C(16x16 f32)
  return __builtin_amdgcn_wmma_f32_16x16x32_bf16(false, a, false, b, (short)0, c, false, false);
}

// A-operand 16x32 fragment from row-major [row, k] (stride ld):
// lane L: row = row0 + (L&15); k = 16*(L>>4)..+15 (one contiguous 32B load)
DEVINL v16bf load_a16(const __bf16* base, int ld, int row0, int k0, int lane) {
  const __bf16* p = base + (size_t)(row0 + (lane & 15)) * ld + k0 + ((lane >> 4) << 4);
  return *(const v16bf*)p;
}

// B-operand 32x16 fragment, element (k,n) = src[n0+n, k0+k], src row-major stride ld:
// lane L: n = L&15, k = 16*(L>>4)..+15 (one contiguous 32B load)
DEVINL v16bf load_b16(const __bf16* base, int ld, int n0, int k0, int lane) {
  const __bf16* p = base + (size_t)(n0 + (lane & 15)) * ld + k0 + ((lane >> 4) << 4);
  return *(const v16bf*)p;
}

// row reductions over the 16 lanes of one D-layout row group (wave32)
DEVINL float rr_max(float t) {
  t = fmaxf(t, __shfl_xor(t, 1, 32));
  t = fmaxf(t, __shfl_xor(t, 2, 32));
  t = fmaxf(t, __shfl_xor(t, 4, 32));
  t = fmaxf(t, __shfl_xor(t, 8, 32));
  return t;
}
DEVINL float rr_sum(float t) {
  t += __shfl_xor(t, 1, 32);
  t += __shfl_xor(t, 2, 32);
  t += __shfl_xor(t, 4, 32);
  t += __shfl_xor(t, 8, 32);
  return t;
}

__global__ void convert_bf16_kernel(const float* __restrict__ src, __bf16* __restrict__ dst, int n) {
  int i = blockIdx.x * blockDim.x + threadIdx.x;
  if (i < n) dst[i] = (__bf16)src[i];
}

// Projections: phiT = x @ w_phi^T, theta = x @ w_theta^T, gT = (x @ w_g^T)^T
__global__ __launch_bounds__(128) void proj_kernel(
    const __bf16* __restrict__ xb,
    const __bf16* __restrict__ wphib, const __bf16* __restrict__ wthetab, const __bf16* __restrict__ wgb,
    __bf16* __restrict__ phiT, __bf16* __restrict__ theta, __bf16* __restrict__ gT) {
  const int lane = threadIdx.x & 31;
  const int wave = threadIdx.x >> 5;
  const int b = blockIdx.y;
  const int z = blockIdx.z;
  const int row0 = blockIdx.x * 64 + wave * 16;
  const __bf16* xa = xb + (size_t)b * T_ * C_;
  const __bf16* w = (z == 0) ? wphib : (z == 1) ? wthetab : wgb;

  v16bf a[8];
#pragma unroll
  for (int kk = 0; kk < 8; ++kk) a[kk] = load_a16(xa, C_, row0, kk * 32, lane);

  for (int nc = 0; nc < 8; ++nc) {
    v8f o = {};
#pragma unroll
    for (int kk = 0; kk < 8; ++kk)
      o = wmma_bf(a[kk], load_b16(w, C_, 16 * nc, 32 * kk, lane), o);
    if (z == 2) {
      int ic = 16 * nc + (lane & 15);
      int t0 = row0 + 8 * (lane >> 4);
      v8bf tmp;
#pragma unroll
      for (int v = 0; v < 8; ++v) tmp[v] = (__bf16)o[v];
      *(v8bf*)(gT + (size_t)b * IC_ * T_ + (size_t)ic * T_ + t0) = tmp;
    } else {
      __bf16* dst = ((z == 0) ? phiT : theta) + (size_t)b * T_ * IC_;
#pragma unroll
      for (int v = 0; v < 8; ++v)
        dst[(size_t)(row0 + v + 8 * (lane >> 4)) * IC_ + 16 * nc + (lane & 15)] = (__bf16)o[v];
    }
  }
}

DEVINL void online_update(const v8f& c, float* m, float* l) {
#pragma unroll
  for (int v = 0; v < 8; ++v) {
    float t = rr_max(c[v]);
    float newm = fmaxf(m[v], t);
    float p = __expf(c[v] - newm);
    float rs = rr_sum(p);
    l[v] = l[v] * __expf(m[v] - newm) + rs;
    m[v] = newm;
  }
}

// Fused double-softmax attention, 4 waves/WG, one 16-row tile per wave.
// K0/K1 (and gT in pass 2) column tiles are staged cooperatively into LDS
// (double-buffered; global->VGPR issued before consume, ds_store after) so the
// workgroup reads each B tile from L2 once instead of 4x.
template <int KD>
__global__ __launch_bounds__(128) void attn_kernel(
    const __bf16* __restrict__ qbase, const __bf16* __restrict__ kbase,
    const __bf16* __restrict__ gT, const __bf16* __restrict__ wmaskb,
    const float* __restrict__ xin, const float* __restrict__ wr,
    float* __restrict__ out, int ibase) {
  constexpr int NK    = KD / 32;
  constexpr int KTILE = 32 * KD;             // elements per input per stage
  constexpr int GTILE = IC_ * 32;            // g stage elements (IC rows x 32 cols)
  constexpr int KCH   = KTILE / (128 * 8);   // v8bf chunks per thread per input
  constexpr int GCH   = GTILE / (128 * 8);

  const int tid  = threadIdx.x;
  const int lane = tid & 31;
  const int wave = tid >> 5;
  const int i  = ibase + blockIdx.y;
  const int j0 = 2 * blockIdx.y;
  const int row0 = blockIdx.x * 64 + wave * 16;

  const __bf16* q0 = qbase + (size_t)j0 * T_ * KD;
  const __bf16* q1 = q0 + (size_t)T_ * KD;
  const __bf16* k0 = kbase + (size_t)j0 * T_ * KD;
  const __bf16* k1 = k0 + (size_t)T_ * KD;
  const __bf16* gp = gT + (size_t)i * IC_ * T_;
  const float wr0 = wr[0], wr1 = wr[1];

  extern __shared__ __bf16 smem[];
  __bf16* kst = smem;                        // 2 bufs x 2 inputs x KTILE
  __bf16* gst = kst + 4 * KTILE;             // 2 bufs x GTILE
  __bf16* myt = gst + 2 * GTILE + wave * 2048;  // per-wave transpose scratch (4KB)

  v8bf r0[KCH], r1[KCH], rg[GCH];

  // K tile rows s0..s0+31 are contiguous in (T,KD) row-major -> flat 16B-chunk copy
  auto ldK = [&](const __bf16* src, int s0, v8bf* r) {
#pragma unroll
    for (int c = 0; c < KCH; ++c)
      r[c] = *(const v8bf*)(src + (size_t)s0 * KD + (size_t)(tid + c * 128) * 8);
  };
  auto stK = [&](__bf16* dst, const v8bf* r) {
#pragma unroll
    for (int c = 0; c < KCH; ++c)
      *(v8bf*)(dst + (tid + c * 128) * 8) = r[c];
  };
  auto ldG = [&](int s0, v8bf* r) {
#pragma unroll
    for (int c = 0; c < GCH; ++c) {
      int idx = tid + c * 128;
      r[c] = *(const v8bf*)(gp + (size_t)(idx >> 2) * T_ + s0 + (idx & 3) * 8);
    }
  };
  auto stG = [&](__bf16* dst, const v8bf* r) {
#pragma unroll
    for (int c = 0; c < GCH; ++c) {
      int idx = tid + c * 128;
      *(v8bf*)(dst + (idx >> 2) * 32 + (idx & 3) * 8) = r[c];
    }
  };

  v16bf qa0[NK], qa1[NK];
#pragma unroll
  for (int kk = 0; kk < NK; ++kk) {
    qa0[kk] = load_a16(q0, KD, row0, kk * 32, lane);
    qa1[kk] = load_a16(q1, KD, row0, kk * 32, lane);
  }

  float m0[8], l0[8], m1[8], l1[8];
#pragma unroll
  for (int v = 0; v < 8; ++v) { m0[v] = -1e30f; l0[v] = 0.f; m1[v] = -1e30f; l1[v] = 0.f; }

  // ---- pass 1: inner-softmax stats (full online max/sum: x.x diag ~ ||x||^2 ~ 256) ----
  ldK(k0, 0, r0); ldK(k1, 0, r1);
  stK(kst, r0);   stK(kst + KTILE, r1);
  for (int s0 = 0; s0 < T_; s0 += 32) {
    const int buf = (s0 >> 5) & 1;
    __syncthreads();
    const bool more = (s0 + 32 < T_);
    if (more) { ldK(k0, s0 + 32, r0); ldK(k1, s0 + 32, r1); }
    const __bf16* kb0 = kst + buf * 2 * KTILE;
    const __bf16* kb1 = kb0 + KTILE;
#pragma unroll
    for (int h = 0; h < 2; ++h) {
      v8f c0 = {}, c1 = {};
#pragma unroll
      for (int kk = 0; kk < NK; ++kk)
        c0 = wmma_bf(qa0[kk], load_b16(kb0, KD, 16 * h, kk * 32, lane), c0);
#pragma unroll
      for (int kk = 0; kk < NK; ++kk)
        c1 = wmma_bf(qa1[kk], load_b16(kb1, KD, 16 * h, kk * 32, lane), c1);
      online_update(c0, m0, l0);
      online_update(c1, m1, l1);
    }
    if (more) {
      __bf16* nb = kst + (buf ^ 1) * 2 * KTILE;
      stK(nb, r0); stK(nb + KTILE, r1);
    }
  }

  float il0[8], il1[8];
#pragma unroll
  for (int v = 0; v < 8; ++v) { il0[v] = wr0 / l0[v]; il1[v] = wr1 / l1[v]; }

  // ---- pass 2: outer softmax + PV. c = w0*p0 + w1*p1 with p in (0,1], so
  // |c| <= |w0|+|w1| (tiny): exp(c) cannot overflow -> no running max, no
  // accumulator rescale needed. ----
  float lc[8];
  v8f acc[8];
#pragma unroll
  for (int v = 0; v < 8; ++v) lc[v] = 0.f;
#pragma unroll
  for (int j = 0; j < 8; ++j) acc[j] = (v8f){};

  __syncthreads();   // pass-1 consumers done before restaging buffer 0
  ldK(k0, 0, r0); ldK(k1, 0, r1); ldG(0, rg);
  stK(kst, r0);   stK(kst + KTILE, r1); stG(gst, rg);

  for (int s0 = 0; s0 < T_; s0 += 32) {
    const int buf = (s0 >> 5) & 1;
    __syncthreads();
    const bool more = (s0 + 32 < T_);
    if (more) { ldK(k0, s0 + 32, r0); ldK(k1, s0 + 32, r1); ldG(s0 + 32, rg); }
    const __bf16* kb0 = kst + buf * 2 * KTILE;
    const __bf16* kb1 = kb0 + KTILE;
    const __bf16* gb  = gst + buf * GTILE;

    v8f p[2];
#pragma unroll
    for (int h = 0; h < 2; ++h) {
      v8f c0 = {}, c1 = {};
#pragma unroll
      for (int kk = 0; kk < NK; ++kk)
        c0 = wmma_bf(qa0[kk], load_b16(kb0, KD, 16 * h, kk * 32, lane), c0);
#pragma unroll
      for (int kk = 0; kk < NK; ++kk)
        c1 = wmma_bf(qa1[kk], load_b16(kb1, KD, 16 * h, kk * 32, lane), c1);
#pragma unroll
      for (int v = 0; v < 8; ++v)
        p[h][v] = __expf(il0[v] * __expf(c0[v] - m0[v]) + il1[v] * __expf(c1[v] - m1[v]));
    }
#pragma unroll
    for (int v = 0; v < 8; ++v) {
      lc[v] += rr_sum(p[0][v] + p[1][v]);
      int r = v + 8 * (lane >> 4);
      myt[r * 32 + (lane & 15)]      = (__bf16)p[0][v];  // 16x32 Y tile, row-major
      myt[r * 32 + 16 + (lane & 15)] = (__bf16)p[1][v];
    }
    v16bf ya = load_a16(myt, 32, 0, 0, lane);            // D-layout -> A-layout
#pragma unroll
    for (int j = 0; j < 8; ++j)
      acc[j] = wmma_bf(ya, load_b16(gb, 32, 16 * j, 0, lane), acc[j]);

    if (more) {
      __bf16* nb = kst + (buf ^ 1) * 2 * KTILE;
      stK(nb, r0); stK(nb + KTILE, r1); stG(gst + (buf ^ 1) * GTILE, rg);
    }
  }

#pragma unroll
  for (int v = 0; v < 8; ++v) {
    float inv = 1.f / lc[v];
#pragma unroll
    for (int j = 0; j < 8; ++j) acc[j][v] *= inv;
  }

  // mtg (16x128) -> LDS as bf16, then out = mtg @ w_mask^T + x^T
#pragma unroll
  for (int j = 0; j < 8; ++j)
#pragma unroll
    for (int v = 0; v < 8; ++v) {
      int r = v + 8 * (lane >> 4);
      myt[r * 128 + 16 * j + (lane & 15)] = (__bf16)acc[j][v];
    }

  v16bf ma[4];
#pragma unroll
  for (int kk = 0; kk < 4; ++kk) ma[kk] = load_a16(myt, 128, 0, kk * 32, lane);

  const float* xr = xin + (size_t)i * T_ * C_;
  float* outp = out + (size_t)i * C_ * T_;
  for (int cc = 0; cc < 16; ++cc) {
    v8f o = {};
#pragma unroll
    for (int kk = 0; kk < 4; ++kk)
      o = wmma_bf(ma[kk], load_b16(wmaskb, IC_, 16 * cc, 32 * kk, lane), o);
    int c = 16 * cc + (lane & 15);
    int t0 = row0 + 8 * (lane >> 4);
    v8f r;
#pragma unroll
    for (int v = 0; v < 8; ++v) r[v] = o[v] + xr[(size_t)(t0 + v) * C_ + c];
    *(v8f*)(outp + (size_t)c * T_ + t0) = r;             // contiguous 32B store
  }
}

extern "C" void kernel_launch(void* const* d_in, const int* in_sizes, int n_in,
                              void* d_out, int out_size, void* d_ws, size_t ws_size,
                              hipStream_t stream) {
  (void)in_sizes; (void)n_in; (void)out_size; (void)ws_size;
  const float* x       = (const float*)d_in[0];
  const float* w_phi   = (const float*)d_in[1];
  const float* w_theta = (const float*)d_in[2];
  const float* w_g     = (const float*)d_in[3];
  const float* w_mask  = (const float*)d_in[4];
  const float* w_r     = (const float*)d_in[5];
  float* out = (float*)d_out;

  // ws layout (bf16 elements), total ~42.2 MB — fits MI455X 192MB L2
  __bf16* ws      = (__bf16*)d_ws;
  __bf16* xb      = ws;                                   // B*T*C
  __bf16* theta   = xb + (size_t)B_ * T_ * C_;            // B*T*IC
  __bf16* phiT    = theta + (size_t)B_ * T_ * IC_;        // B*T*IC
  __bf16* gT      = phiT + (size_t)B_ * T_ * IC_;         // B*IC*T
  __bf16* wphib   = gT + (size_t)B_ * IC_ * T_;           // IC*C
  __bf16* wthetab = wphib + (size_t)IC_ * C_;
  __bf16* wgb     = wthetab + (size_t)IC_ * C_;
  __bf16* wmaskb  = wgb + (size_t)IC_ * C_;               // C*IC

  int n = B_ * T_ * C_;
  convert_bf16_kernel<<<(n + 255) / 256, 256, 0, stream>>>(x, xb, n);
  n = IC_ * C_;
  convert_bf16_kernel<<<(n + 255) / 256, 256, 0, stream>>>(w_phi, wphib, n);
  convert_bf16_kernel<<<(n + 255) / 256, 256, 0, stream>>>(w_theta, wthetab, n);
  convert_bf16_kernel<<<(n + 255) / 256, 256, 0, stream>>>(w_g, wgb, n);
  n = C_ * IC_;
  convert_bf16_kernel<<<(n + 255) / 256, 256, 0, stream>>>(w_mask, wmaskb, n);

  proj_kernel<<<dim3(T_ / 64, B_, 3), 128, 0, stream>>>(xb, wphib, wthetab, wgb, phiT, theta, gT);

  // dynamic LDS: stage buffers (2 x 2 x 32*KD) + g stage (2 x IC*32) + 4x4KB scratch
  size_t sh256 = (size_t)(4 * 32 * 256 + 2 * IC_ * 32 + 4 * 2048) * 2;  // 96 KB
  size_t sh128 = (size_t)(4 * 32 * 128 + 2 * IC_ * 32 + 4 * 2048) * 2;  // 64 KB

  // i = 0..7 : maps from raw x (K=256), paired batches (2i, 2i+1)
  attn_kernel<256><<<dim3(T_ / 64, 8), 128, sh256, stream>>>(xb, xb, gT, wmaskb, x, w_r, out, 0);
  // i = 8..15: maps from theta @ phi (K=128), paired batches (2(i-8), 2(i-8)+1)
  attn_kernel<128><<<dim3(T_ / 64, 8), 128, sh128, stream>>>(theta, phiT, gT, wmaskb, x, w_r, out, 8);
}
